// Model_12386685681908
// MI455X (gfx1250) — matrix operand order repaired and checked by reading, not compiled, not run on hardware
//
#include <hip/hip_runtime.h>

typedef __attribute__((ext_vector_type(16))) _Float16 v16h;
typedef __attribute__((ext_vector_type(8)))  _Float16 v8h;
typedef __attribute__((ext_vector_type(4)))  _Float16 v4h;
typedef __attribute__((ext_vector_type(8)))  float    v8f;

#define BATCH 8
#define KNN 32
#define PCH 64
#define APITCH 40   // LDS row pitch in halves: 80B = 20 dwords -> conflict-free b128 frag reads

// ---------------------------------------------------------------------------
// Weight pre-conversion: W[M][Kc] f32 -> Wh[M][Kpad] f16, zero padded K tail.
// ---------------------------------------------------------------------------
__global__ void convert_w_kernel(const float* __restrict__ Wm, _Float16* __restrict__ Wh,
                                 int Kc, int Kpad, long total) {
    long t = (long)blockIdx.x * 256 + threadIdx.x;
    if (t >= total) return;
    int c = (int)(t % Kpad);
    long r = t / Kpad;
    Wh[t] = (c < Kc) ? (_Float16)Wm[r * Kc + c] : (_Float16)0.f;
}

// ---------------------------------------------------------------------------
// WMMA GEMM:  Y[M, Ncols] = Wh[M, Kpad](f16) * X[Kpad, Ncols](f32, rows >= Kc
// zeroed by producer).  M, Ncols multiples of 64; Kpad multiple of 32.
// Block = 128 threads (4 waves), tile 64x64, each wave 32x32 (2x2 WMMA).
// A staged [m][k]; B staged TRANSPOSED [n][k] so each lane's 16 halves are
// two 16-byte k runs (k 8h.., 16+8h..) -> two ds_load_b128 per fragment. Branch-free.
// ---------------------------------------------------------------------------
__global__ __launch_bounds__(128)
void wmma_gemm(const _Float16* __restrict__ Wh, const float* __restrict__ Xm,
               float* __restrict__ Y, int Kpad, long Ncols) {
    __shared__ _Float16 As[64][APITCH];
    __shared__ _Float16 Bt[64][APITCH];

    const int tid  = threadIdx.x;
    const int wave = tid >> 5;
    const int lane = tid & 31;
    const int m0   = blockIdx.y * 64;
    const long n0  = (long)blockIdx.x * 64;
    const int wm   = (wave >> 1) * 32;
    const int wn   = (wave & 1) * 32;
    const int lm   = lane & 15;
    const int lh   = lane >> 4;

    v8f acc[2][2] = {};

    for (int k0 = 0; k0 < Kpad; k0 += 32) {
        #pragma unroll
        for (int p = 0; p < 2; ++p) {               // A: 64x32 halves, b128 in/out
            int r  = (tid >> 2) + p * 32;
            int c8 = (tid & 3) * 8;
            *(v8h*)&As[r][c8] = *(const v8h*)&Wh[(long)(m0 + r) * Kpad + k0 + c8];
        }
        #pragma unroll
        for (int p = 0; p < 4; ++p) {               // B: 4 k-rows/pass, coalesced b32
            int r = (tid >> 6) * 16 + p * 4;
            int c = tid & 63;
            const float* src = &Xm[(long)(k0 + r) * Ncols + n0 + c];
            v4h h = { (_Float16)src[0], (_Float16)src[Ncols],
                      (_Float16)src[2 * Ncols], (_Float16)src[3 * Ncols] };
            *(v4h*)&Bt[c][r] = h;                   // ds_store_b64
        }
        __syncthreads();

        v16h af[2], bf[2];
        #pragma unroll
        for (int t = 0; t < 2; ++t) {
            const int row = wm + t * 16 + lm;           // ISA A 16x32 layout
            v8h a0 = *(const v8h*)&As[row][lh * 8];
            v8h a1 = *(const v8h*)&As[row][16 + lh * 8];
            const int col = wn + t * 16 + lm;           // ISA B 32x16 layout
            v8h b0 = *(const v8h*)&Bt[col][lh * 8];
            v8h b1 = *(const v8h*)&Bt[col][16 + lh * 8];
            #pragma unroll
            for (int e = 0; e < 8; ++e) {
                af[t][e] = a0[e]; af[t][8 + e] = a1[e];
                bf[t][e] = b0[e]; bf[t][8 + e] = b1[e];
            }
        }
        #pragma unroll
        for (int i = 0; i < 2; ++i)
            #pragma unroll
            for (int j = 0; j < 2; ++j)
                acc[i][j] = __builtin_amdgcn_wmma_f32_16x16x32_f16(
                    false, af[i], false, bf[j], (short)0, acc[i][j], false, false);
        __syncthreads();
    }

    // D layout: reg r -> M = r + 8*lh ; N = lm
    #pragma unroll
    for (int i = 0; i < 2; ++i)
        #pragma unroll
        for (int j = 0; j < 2; ++j) {
            int mbase = m0 + wm + i * 16 + lh * 8;
            long nidx = n0 + wn + j * 16 + lm;
            #pragma unroll
            for (int r = 0; r < 8; ++r)
                Y[(long)(mbase + r) * Ncols + nidx] = acc[i][j][r];
        }
}

// ---------------------------------------------------------------------------
// BatchNorm (training stats over all columns), deterministic 3-stage.
// ---------------------------------------------------------------------------
__global__ __launch_bounds__(256)
void bn_partial(const float* __restrict__ y, float* __restrict__ part, long Ncols) {
    int c = blockIdx.y, p = blockIdx.x;
    long per = (Ncols + PCH - 1) / PCH;
    long s0 = (long)p * per;
    long s1 = s0 + per; if (s1 > Ncols) s1 = Ncols;
    const float* row = y + (long)c * Ncols;
    float sum = 0.f, sq = 0.f;
    for (long i = s0 + threadIdx.x; i < s1; i += 256) {
        float v = row[i]; sum += v; sq += v * v;
    }
    __shared__ float rs[256], rq[256];
    rs[threadIdx.x] = sum; rq[threadIdx.x] = sq;
    __syncthreads();
    for (int st = 128; st > 0; st >>= 1) {
        if (threadIdx.x < st) { rs[threadIdx.x] += rs[threadIdx.x + st];
                                rq[threadIdx.x] += rq[threadIdx.x + st]; }
        __syncthreads();
    }
    if (threadIdx.x == 0) {
        part[((long)c * PCH + p) * 2]     = rs[0];
        part[((long)c * PCH + p) * 2 + 1] = rq[0];
    }
}

__global__ __launch_bounds__(PCH)
void bn_finalize(const float* __restrict__ part, float* __restrict__ ms, long Ncols) {
    int c = blockIdx.x, t = threadIdx.x;
    __shared__ float rs[PCH], rq[PCH];
    rs[t] = part[((long)c * PCH + t) * 2];
    rq[t] = part[((long)c * PCH + t) * 2 + 1];
    __syncthreads();
    for (int st = PCH / 2; st > 0; st >>= 1) {
        if (t < st) { rs[t] += rs[t + st]; rq[t] += rq[t + st]; }
        __syncthreads();
    }
    if (t == 0) {
        float mean = rs[0] / (float)Ncols;
        float var  = rq[0] / (float)Ncols - mean * mean;
        ms[c * 2]     = mean;
        ms[c * 2 + 1] = rsqrtf(var + 1e-5f);
    }
}

__global__ __launch_bounds__(256)
void bn_apply(float* __restrict__ y, const float* __restrict__ ms,
              const float* __restrict__ res, long Ncols, int act) {
    int c = blockIdx.y;
    long i = (long)blockIdx.x * 256 + threadIdx.x;
    if (i >= Ncols) return;
    long o = (long)c * Ncols + i;
    float v = (y[o] - ms[c * 2]) * ms[c * 2 + 1];
    if (res) v += res[o];
    if (act) v = fmaxf(v, 0.f);
    y[o] = v;
}

// ---------------------------------------------------------------------------
// Misc small kernels
// ---------------------------------------------------------------------------
__global__ void xyz_from_x_kernel(const float* __restrict__ x, float* __restrict__ xyz, int N) {
    long t = (long)blockIdx.x * 256 + threadIdx.x;
    if (t >= (long)BATCH * N) return;
    int b = (int)(t / N), n = (int)(t % N);
    const float* xb = x + (long)b * 3 * N;
    xyz[t * 3 + 0] = xb[n];
    xyz[t * 3 + 1] = xb[N + n];
    xyz[t * 3 + 2] = xb[2L * N + n];
}

__global__ void embed_kernel(const float* __restrict__ x, const float* __restrict__ W,
                             float* __restrict__ y, int N) {
    int c = blockIdx.y;
    long col = (long)blockIdx.x * 256 + threadIdx.x;  // b*N+n
    if (col >= (long)BATCH * N) return;
    int b = (int)(col / N), n = (int)(col % N);
    const float* xb = x + (long)b * 3 * N + n;
    float v = W[c * 3 + 0] * xb[0] + W[c * 3 + 1] * xb[N] + W[c * 3 + 2] * xb[2L * N];
    y[(long)c * BATCH * N + col] = v;
}

// Furthest point sampling, one block per batch (matches jax scan semantics)
__global__ __launch_bounds__(256)
void fps_kernel(const float* __restrict__ xyz, int N, int S, int* __restrict__ idx) {
    __shared__ float dist[2048];
    __shared__ float rv[256];
    __shared__ int   ri[256];
    __shared__ int   s_last;
    int b = blockIdx.x, t = threadIdx.x;
    for (int i = t; i < N; i += 256) dist[i] = 1e10f;
    if (t == 0) s_last = 0;
    __syncthreads();
    for (int s = 0; s < S; ++s) {
        int last = s_last;
        if (t == 0) idx[b * S + s] = last;
        long lb = ((long)b * N + last) * 3;
        float lx = xyz[lb], ly = xyz[lb + 1], lz = xyz[lb + 2];
        float bv = -1.f; int bi = 0;
        for (int i = t; i < N; i += 256) {
            long o = ((long)b * N + i) * 3;
            float dx = xyz[o] - lx, dy = xyz[o + 1] - ly, dz = xyz[o + 2] - lz;
            float dd = dx * dx + dy * dy + dz * dz;
            float d = dist[i]; d = fminf(d, dd); dist[i] = d;
            if (d > bv) { bv = d; bi = i; }
        }
        rv[t] = bv; ri[t] = bi;
        __syncthreads();
        for (int st = 128; st > 0; st >>= 1) {
            if (t < st) {
                if (rv[t + st] > rv[t] || (rv[t + st] == rv[t] && ri[t + st] < ri[t])) {
                    rv[t] = rv[t + st]; ri[t] = ri[t + st];
                }
            }
            __syncthreads();
        }
        if (t == 0) s_last = ri[0];
        __syncthreads();
    }
}

__global__ void gather_new_kernel(const float* __restrict__ f, const float* __restrict__ xyz,
                                  const int* __restrict__ fps_idx, float* __restrict__ new_xyz,
                                  float* __restrict__ new_pts, int Nprev, int S, int last) {
    int s = blockIdx.x, b = blockIdx.y, c = threadIdx.x;  // blockDim.x == last
    int gi = fps_idx[b * S + s];
    new_pts[((long)b * S + s) * last + c] =
        f[(long)c * ((long)BATCH * Nprev) + (long)b * Nprev + gi];
    if (c < 3)
        new_xyz[((long)b * S + s) * 3 + c] = xyz[((long)b * Nprev + gi) * 3 + c];
}

// kNN: 32 smallest distances, tie -> lowest index (matches lax.top_k on -dist)
__global__ __launch_bounds__(128)
void knn_kernel(const float* __restrict__ nxyz, const float* __restrict__ xyz,
                int N, int S, int* __restrict__ knn) {
    __shared__ float d[2048];
    __shared__ float rv[128];
    __shared__ int   ri[128];
    int s = blockIdx.x, b = blockIdx.y, t = threadIdx.x;
    long cb = ((long)b * S + s) * 3;
    float cx = nxyz[cb], cy = nxyz[cb + 1], cz = nxyz[cb + 2];
    for (int i = t; i < N; i += 128) {
        long o = ((long)b * N + i) * 3;
        float dx = xyz[o] - cx, dy = xyz[o + 1] - cy, dz = xyz[o + 2] - cz;
        d[i] = dx * dx + dy * dy + dz * dz;
    }
    __syncthreads();
    for (int k = 0; k < KNN; ++k) {
        float bv = 3.4e38f; int bi = N;
        for (int i = t; i < N; i += 128) {
            float v = d[i];
            if (v < bv) { bv = v; bi = i; }
        }
        rv[t] = bv; ri[t] = bi;
        __syncthreads();
        for (int st = 64; st > 0; st >>= 1) {
            if (t < st) {
                if (rv[t + st] < rv[t] || (rv[t + st] == rv[t] && ri[t + st] < ri[t])) {
                    rv[t] = rv[t + st]; ri[t] = ri[t + st];
                }
            }
            __syncthreads();
        }
        if (t == 0) { knn[((long)b * S + s) * KNN + k] = ri[0]; d[ri[0]] = 3.39e38f; }
        __syncthreads();
    }
}

// Gather neighbors, subtract per-(b,s,ch) mean over K, write centered into FEAT,
// accumulate sum-of-squares per (b,s). One 32-thread block per (s,b); lane = k.
__global__ __launch_bounds__(32)
void group_center_kernel(const float* __restrict__ f, const float* __restrict__ xyz,
                         const int* __restrict__ knn, float* __restrict__ feat,
                         float* __restrict__ ssq, int Nprev, int S, int last, long NcK) {
    __shared__ float red[32];
    int s = blockIdx.x, b = blockIdx.y, k = threadIdx.x;
    int j = knn[((long)b * S + s) * KNN + k];
    long colbase = ((long)b * S + s) * KNN;
    float acc = 0.f;
    int d3 = last + 3;
    for (int c = 0; c < d3; ++c) {
        float v;
        if (c < last) v = f[(long)c * ((long)BATCH * Nprev) + (long)b * Nprev + j];
        else          v = xyz[((long)b * Nprev + j) * 3 + (c - last)];
        red[k] = v; __syncthreads();
        for (int st = 16; st > 0; st >>= 1) {
            if (k < st) red[k] += red[k + st];
            __syncthreads();
        }
        float mean = red[0] * (1.f / 32.f);
        __syncthreads();
        float cv = v - mean;
        feat[(long)c * NcK + colbase + k] = cv;
        acc += cv * cv;
    }
    red[k] = acc; __syncthreads();
    for (int st = 16; st > 0; st >>= 1) {
        if (k < st) red[k] += red[k + st];
        __syncthreads();
    }
    if (k == 0) ssq[(long)b * S + s] = red[0];
}

__global__ __launch_bounds__(256)
void batch_std_kernel(const float* __restrict__ ssq, float* __restrict__ inv, int S, long cnt) {
    int b = blockIdx.x;
    __shared__ float r[256];
    float a = 0.f;
    for (int i = threadIdx.x; i < S; i += 256) a += ssq[(long)b * S + i];
    r[threadIdx.x] = a; __syncthreads();
    for (int st = 128; st > 0; st >>= 1) {
        if (threadIdx.x < st) r[threadIdx.x] += r[threadIdx.x + st];
        __syncthreads();
    }
    if (threadIdx.x == 0) {
        float sd = sqrtf(r[0] / (float)(cnt - 1));  // ddof=1, mean of centered == 0
        inv[b] = 1.f / (sd + 1e-5f);
    }
}

// Rows [0,last+3): affine-scale centered values; [last+3, 2last+3): broadcast
// sampled features; [2last+3, Kpad): zero fill (K padding for the WMMA GEMM).
__global__ __launch_bounds__(256)
void feat_finalize_kernel(float* __restrict__ feat, const float* __restrict__ inv,
                          const float* __restrict__ alpha, const float* __restrict__ beta,
                          const float* __restrict__ new_pts, int S, int last, long NcK) {
    int c = blockIdx.y;
    long n = (long)blockIdx.x * 256 + threadIdx.x;
    if (n >= NcK) return;
    int d3 = last + 3;
    long o = (long)c * NcK + n;
    if (c < d3) {
        int b = (int)(n / ((long)S * KNN));
        feat[o] = feat[o] * inv[b] * alpha[c] + beta[c];
    } else if (c < d3 + last) {
        long sb = n >> 5;  // (b*S+s)
        feat[o] = new_pts[sb * last + (c - d3)];
    } else {
        feat[o] = 0.f;     // zero-padded K rows
    }
}

__global__ __launch_bounds__(64)
void maxpool_kernel(const float* __restrict__ y, float* __restrict__ f, long NcS) {
    int c = blockIdx.y;
    long col = (long)blockIdx.x * 64 + threadIdx.x;
    if (col >= NcS) return;
    const float* p = y + (long)c * (NcS * KNN) + col * KNN;
    float m = p[0];
    #pragma unroll
    for (int k = 1; k < KNN; ++k) m = fmaxf(m, p[k]);
    f[(long)c * NcS + col] = m;
}

__global__ void out_kernel(const float* __restrict__ f, float* __restrict__ out, int C, int S) {
    long t = (long)blockIdx.x * 256 + threadIdx.x;
    long total = (long)BATCH * C * S;
    if (t >= total) return;
    int s = (int)(t % S);
    long r = t / S;
    int c = (int)(r % C), b = (int)(r / C);
    out[t] = f[(long)c * ((long)BATCH * S) + (long)b * S + s];
}

// ---------------------------------------------------------------------------
// Host orchestration
// ---------------------------------------------------------------------------
extern "C" void kernel_launch(void* const* d_in, const int* in_sizes, int n_in,
                              void* d_out, int out_size, void* d_ws, size_t ws_size,
                              hipStream_t stream) {
    (void)in_sizes; (void)n_in; (void)out_size; (void)ws_size;

    const float* X    = (const float*)d_in[0];
    const float* embW = (const float*)d_in[1];
    auto P = [&](int i, int j) { return (const float*)d_in[3 + 20 * i + j]; };
    // j: 0 alpha, 1 beta, 2 trW, 3 trb, 4..7 pre0, 8..11 pre1, 12..15 pos0, 16..19 pos1
    // (all conv biases are absorbed by training-mode BN mean subtraction -> unused)

    float* w = (float*)d_ws;
    size_t off = 0;
    auto alloc = [&](size_t nfl) { float* p = w + off; off += (nfl + 255) & ~(size_t)255; return p; };

    float* FEAT = alloc(41943040);   // 160 rows * (B*1024*32); K-padded, also 3rd GEMM buf
    float* A    = alloc(33554432);
    float* Bb   = alloc(33554432);
    float* F0   = alloc(1048576);
    float* F1   = alloc(1048576);
    float* F2   = alloc(1048576);
    float* XYZa = alloc(49152);
    float* XYZb = alloc(49152);
    float* NP   = alloc(524288);
    float* SSQ  = alloc(8192);
    float* INV  = alloc(256);
    float* PART = alloc((size_t)1024 * PCH * 2);
    float* MS   = alloc(2048);
    _Float16* WH = (_Float16*)alloc(550912);   // >= 1024*1056 halves
    int*   FPSI = (int*)alloc(8192);
    int*   KNNI = (int*)alloc(262144);

    auto bn = [&](float* y, const float* res, int C, long Ncols, int act) {
        dim3 g1(PCH, C);
        bn_partial<<<g1, 256, 0, stream>>>(y, PART, Ncols);
        bn_finalize<<<C, PCH, 0, stream>>>(PART, MS, Ncols);
        dim3 g2((unsigned)((Ncols + 255) / 256), C);
        bn_apply<<<g2, 256, 0, stream>>>(y, MS, res, Ncols, act);
    };
    auto gemm = [&](const float* Wm, const float* Xm, float* Y, int M, int Kc, long Ncols) {
        int Kpad = (Kc + 31) & ~31;
        long tot = (long)M * Kpad;
        convert_w_kernel<<<(unsigned)((tot + 255) / 256), 256, 0, stream>>>(Wm, WH, Kc, Kpad, tot);
        dim3 g((unsigned)(Ncols / 64), M / 64);
        wmma_gemm<<<g, 128, 0, stream>>>(WH, Xm, Y, Kpad, Ncols);
    };

    // ---- embedding ----
    int N = 2048;
    {
        long tot = (long)BATCH * N;
        xyz_from_x_kernel<<<(unsigned)((tot + 255) / 256), 256, 0, stream>>>(X, XYZa, N);
        dim3 ge((unsigned)((tot + 255) / 256), 64);
        embed_kernel<<<ge, 256, 0, stream>>>(X, embW, F0, N);
        bn(F0, nullptr, 64, tot, 1);
    }

    float* fcur = F0; float* fs1 = F1; float* fs2 = F2;
    float* xyzc = XYZa; float* xyzn = XYZb;
    int last = 64;

    for (int i = 0; i < 4; ++i) {
        int S = N / 2;
        int out = 2 * last;
        int d = 2 * last + 3;
        int Kpad = (d + 31) & ~31;
        long NcK = (long)BATCH * S * KNN;
        long NcS = (long)BATCH * S;

        // ---- LocalGrouper ----
        fps_kernel<<<BATCH, 256, 0, stream>>>(xyzc, N, S, FPSI);
        gather_new_kernel<<<dim3(S, BATCH), last, 0, stream>>>(fcur, xyzc, FPSI, xyzn, NP, N, S, last);
        knn_kernel<<<dim3(S, BATCH), 128, 0, stream>>>(xyzn, xyzc, N, S, KNNI);
        group_center_kernel<<<dim3(S, BATCH), 32, 0, stream>>>(fcur, xyzc, KNNI, FEAT, SSQ, N, S, last, NcK);
        batch_std_kernel<<<BATCH, 256, 0, stream>>>(SSQ, INV, S, (long)S * KNN * (last + 3));
        feat_finalize_kernel<<<dim3((unsigned)(NcK / 256), Kpad), 256, 0, stream>>>(
            FEAT, INV, P(i, 0), P(i, 1), NP, S, last, NcK);

        // ---- PreExtraction (WMMA GEMMs over B*S*K columns) ----
        gemm(P(i, 2), FEAT, A, out, d, NcK);            bn(A, nullptr, out, NcK, 1);   // transfer CBR
        gemm(P(i, 4), A, Bb, out, out, NcK);            bn(Bb, nullptr, out, NcK, 1);  // pre0 conv1
        gemm(P(i, 6), Bb, FEAT, out, out, NcK);         bn(FEAT, A, out, NcK, 1);      // pre0 conv2 + res
        gemm(P(i, 8), FEAT, Bb, out, out, NcK);         bn(Bb, nullptr, out, NcK, 1);  // pre1 conv1
        gemm(P(i, 10), Bb, A, out, out, NcK);           bn(A, FEAT, out, NcK, 1);      // pre1 conv2 + res
        maxpool_kernel<<<dim3((unsigned)((NcS + 63) / 64), out), 64, 0, stream>>>(A, fs1, NcS);

        // ---- PosExtraction (WMMA GEMMs over B*S columns) ----
        gemm(P(i, 12), fs1, fs2, out, out, NcS);        bn(fs2, nullptr, out, NcS, 1); // pos0 conv1
        gemm(P(i, 14), fs2, fcur, out, out, NcS);       bn(fcur, fs1, out, NcS, 1);    // pos0 conv2 + res
        gemm(P(i, 16), fcur, fs2, out, out, NcS);       bn(fs2, nullptr, out, NcS, 1); // pos1 conv1
        gemm(P(i, 18), fs2, fs1, out, out, NcS);        bn(fs1, fcur, out, NcS, 1);    // pos1 conv2 + res

        // rotate buffers
        float* t = fcur; fcur = fs1; fs1 = fs2; fs2 = t;
        float* tx = xyzc; xyzc = xyzn; xyzn = tx;
        N = S; last = out;
    }

    long total = (long)BATCH * 1024 * 128;
    out_kernel<<<(unsigned)((total + 255) / 256), 256, 0, stream>>>(fcur, (float*)d_out, 1024, 128);
}
